// rnn_model_61435212202131
// MI455X (gfx1250) — matrix operand order repaired and checked by reading, not compiled, not run on hardware
//
#include <hip/hip_runtime.h>

// ---------------------------------------------------------------------------
// Model dims
// ---------------------------------------------------------------------------
#define BATCH 128
#define T_LEN 1000
#define HDIM  256
#define LROW  264          // padded LDS row pitch (bf16 elements) -> conflict-free
#define FC_K  256000       // T_LEN * HDIM
#define FC_N  512
#define KSPLIT 32
#define KCHUNK 8000        // FC_K / KSPLIT = 250 k-steps of 32

typedef __attribute__((ext_vector_type(16))) __bf16 v16bf;
typedef __attribute__((ext_vector_type(8)))  __bf16 v8bf;
typedef __attribute__((ext_vector_type(8)))  float  v8f;
typedef __attribute__((ext_vector_type(4)))  unsigned int v4u;
typedef __attribute__((ext_vector_type(8)))  int v8i;
typedef __attribute__((ext_vector_type(4)))  int v4i;

// TDM builtin arity differs between toolchains; the gfx1250 TDM header only
// ships with the 6-arg (therock) toolchain -> use it as the discriminator.
#if defined(__has_include)
#if __has_include(<hip/amd_detail/amd_gfx1250_TDM.h>)
#define TDM_SIX_ARGS 1
#else
#define TDM_SIX_ARGS 0
#endif
#else
#define TDM_SIX_ARGS 0
#endif

// ---------------------------------------------------------------------------
// Helpers
// ---------------------------------------------------------------------------
static __device__ inline v8f wmma_bf16(v16bf a, v16bf b, v8f c) {
  // D = A(16x32 bf16) * B(32x16 bf16) + C(16x16 f32)
  return __builtin_amdgcn_wmma_f32_16x16x32_bf16(false, a, false, b,
                                                 (short)0, c, false, false);
}

static __device__ inline v16bf mk16(v8bf lo, v8bf hi) {
  v16bf r;
#pragma unroll
  for (int i = 0; i < 8; ++i) { r[i] = lo[i]; r[i + 8] = hi[i]; }
  return r;
}

// A-operand fragment per ISA layout: lane<16 -> K[kh..kh+8) and K[kh+16..kh+24)
// with kh = c*32 + (lane>=16 ? 8 : 0); rowp points at this lane's M-row.
static __device__ inline v16bf lds_loadA(const __bf16* rowp, int kh) {
  v8bf lo = *(const v8bf*)(rowp + kh);
  v8bf hi = *(const v8bf*)(rowp + kh + 16);
  return mk16(lo, hi);
}

// Load f32 K[0..8) and K[16..24) from p and convert to a bf16 B-fragment (table order).
static __device__ inline v16bf cvt16(const float4* p) {
  float4 q0 = p[0], q1 = p[1], q2 = p[4], q3 = p[5];
  v16bf r;
  r[0]=(__bf16)q0.x; r[1]=(__bf16)q0.y; r[2]=(__bf16)q0.z; r[3]=(__bf16)q0.w;
  r[4]=(__bf16)q1.x; r[5]=(__bf16)q1.y; r[6]=(__bf16)q1.z; r[7]=(__bf16)q1.w;
  r[8]=(__bf16)q2.x; r[9]=(__bf16)q2.y; r[10]=(__bf16)q2.z; r[11]=(__bf16)q2.w;
  r[12]=(__bf16)q3.x; r[13]=(__bf16)q3.y; r[14]=(__bf16)q3.z; r[15]=(__bf16)q3.w;
  return r;
}

// ---- Tensor Data Mover: raw descriptor call (groups 2/3 zero: <=2D tiles) ---
static __device__ inline void tdm_call(v4u g0, v8i g1) {
  v4i z4 = (v4i){0, 0, 0, 0};
#if TDM_SIX_ARGS
  v8i z8 = (v8i){0, 0, 0, 0, 0, 0, 0, 0};
  __builtin_amdgcn_tensor_load_to_lds(g0, g1, z4, z4, z8, 0);
#else
  __builtin_amdgcn_tensor_load_to_lds(g0, g1, z4, z4, 0);
#endif
}

// 2D f32 tile: tile_d1 rows of tile_d0 elements, row stride stride0 elements,
// packed contiguously into LDS at lds_off.
static __device__ inline void tdm_load_f32_2d(unsigned lds_off, const float* g,
                                              unsigned tile_d0, unsigned tile_d1,
                                              unsigned stride0) {
  unsigned long long ga = (unsigned long long)(const void*)g;
  v4u g0;
  g0.x = 1u;                                   // count=1 (user descriptor)
  g0.y = lds_off;                              // LDS byte address
  g0.z = (unsigned)ga;                         // global addr lo
  g0.w = (unsigned)((ga >> 32) & 0x01FFFFFFu) | 0x80000000u;  // addr hi | type=2
  v8i g1;
  g1[0] = (int)(2u << 16);                     // data_size=4B
  g1[1] = (int)(tile_d0 << 16);                // tensor_dim0 lo16
  g1[2] = (int)((tile_d0 >> 16) | (tile_d1 << 16)); // dim0 hi | tensor_dim1 lo
  g1[3] = (int)((tile_d1 >> 16) | (tile_d0 << 16)); // dim1 hi | tile_dim0
  g1[4] = (int)tile_d1;                        // tile_dim1 (tile_dim2=0)
  g1[5] = (int)stride0;                        // tensor_dim0_stride lo32
  g1[6] = 0;                                   // stride hi | dim1_stride lo
  g1[7] = 0;
  tdm_call(g0, g1);
}

// 1D bf16 block of n elements, with LDS padding: +4 DWORDs after every
// 128 DWORDs (256 halves -> LROW=264 pitch).
static __device__ inline void tdm_load_bf16_padded(unsigned lds_off,
                                                   const __bf16* g, unsigned n) {
  unsigned long long ga = (unsigned long long)(const void*)g;
  v4u g0;
  g0.x = 1u;
  g0.y = lds_off;
  g0.z = (unsigned)ga;
  g0.w = (unsigned)((ga >> 32) & 0x01FFFFFFu) | 0x80000000u;
  v8i g1;
  // data_size=2B | pad_enable | pad_interval=6 (128 DW) | pad_amount=3 (4 DW)
  g1[0] = (int)((1u << 16) | (1u << 20) | (6u << 22) | (3u << 25));
  g1[1] = (int)(n << 16);                      // tensor_dim0 lo16
  g1[2] = (int)((n >> 16) | (1u << 16));       // dim0 hi | tensor_dim1=1
  g1[3] = (int)(n << 16);                      // tile_dim0 (n <= 65535)
  g1[4] = 0;                                   // tile_dim1=0 (1D)
  g1[5] = (int)n;                              // stride (unused for 1D)
  g1[6] = 0;
  g1[7] = 0;
  tdm_call(g0, g1);
}

static __device__ inline unsigned lds_addr_of(const void* p) {
  return (unsigned)(unsigned long long)p;      // LDS aperture: addr[31:0]
}

// Cooperative copy of the 16x256 bf16 h-state tile (LDS) to global hseq[.,t,.]
static __device__ inline void copy_h(const __bf16* sH, __bf16* hseq,
                                     int b0, int t, int tid) {
  int r  = tid >> 4;            // 0..15 rows
  int c0 = (tid & 15) * 16;     // halves
  const uint4* ps = (const uint4*)(sH + r * LROW + c0);
  uint4 q0 = ps[0], q1 = ps[1];
  uint4* pd = (uint4*)(hseq + ((long)(b0 + r) * T_LEN + t) * HDIM + c0);
  pd[0] = q0; pd[1] = q1;
}

// ---------------------------------------------------------------------------
// Kernel 1/3: one RNN layer. 8 WGs x 256 threads; each WG owns 16 batch rows.
// W_hh held entirely in VGPRs (bf16 B-fragments) for all 1000 steps.
// layer==0: xp = input[b][t]*W_ih0[n] + (b_ih0+b_hh0)[n]
// layer==1: xp tile DMA'd one step ahead by the TDM into a double buffer.
// ---------------------------------------------------------------------------
__global__ __launch_bounds__(256) void rnn_layer_kernel(
    const float* __restrict__ W_hh,     // (H,H) f32
    const float* __restrict__ h0,       // (B,H) f32 slice for this layer
    const float* __restrict__ input,    // (B,T) f32  (layer 0 only)
    const float* __restrict__ W_ih,     // (H)   f32  (layer 0 only)
    const float* __restrict__ b_a,      // (H)         (layer 0 only)
    const float* __restrict__ b_b,      // (H)         (layer 0 only)
    const float* __restrict__ xp,       // (B*T,H) f32 (layer 1 only)
    __bf16* __restrict__ hseq,          // (B*T,H) bf16 out
    int layer)
{
  __shared__ __bf16 sH[16 * LROW];      // current hidden state, bf16
  __shared__ float  sX[2][16 * 256];    // layer0: input chunk; layer1: xp tiles

  const int tid  = threadIdx.x;
  const int lane = tid & 31;
  const int w    = tid >> 5;            // wave 0..7
  const int lm   = lane & 15;
  const int l16  = lane >> 4;           // 0/1
  const int b0   = blockIdx.x * 16;

  // ---- preload B operand: B[k][n] = W_hh[n][k]; per lane N=lm, two runs of 8 K (table order).
  v16bf Breg[2][8];
  float wihv[2]  = {0.f, 0.f};
  float biasv[2] = {0.f, 0.f};
#pragma unroll
  for (int j = 0; j < 2; ++j) {
    int n = w * 32 + j * 16 + lm;
#pragma unroll
    for (int c = 0; c < 8; ++c) {
      const float4* p = (const float4*)(W_hh + (long)n * HDIM + c * 32 + l16 * 8);
      Breg[j][c] = cvt16(p);
    }
    if (layer == 0) {
      wihv[j]  = W_ih[n];
      biasv[j] = b_a[n] + b_b[n];
    }
  }

  // ---- init h state from h0 (f32 -> bf16)
  {
    int r = tid >> 4, c0 = (tid & 15) * 16;
#pragma unroll
    for (int i = 0; i < 16; ++i)
      sH[r * LROW + c0 + i] = (__bf16)h0[(long)(b0 + r) * HDIM + c0 + i];
  }

  // layer 1: async-prefetch the t=0 xp tile (16 rows, stride T*H) into buf 0
  if (layer == 1 && w == 0) {
    tdm_load_f32_2d(lds_addr_of(&sX[0][0]),
                    xp + (long)b0 * T_LEN * HDIM, HDIM, 16,
                    (unsigned)(T_LEN * HDIM));
  }

  for (int t = 0; t < T_LEN; ++t) {
    // ---- stage input chunk (layer 0 only; every 256 steps)
    if (layer == 0 && (t & 255) == 0) {
      __syncthreads();                       // prior epilogue read sX
      int r = tid >> 4, c0 = (tid & 15) * 16;
#pragma unroll
      for (int i = 0; i < 16; ++i) {
        int tt = t + c0 + i;
        sX[0][r * 256 + c0 + i] = (tt < T_LEN)
            ? input[(long)(b0 + r) * T_LEN + tt] : 0.f;
      }
    }
    __syncthreads();   // barrier A: sH(t) ready; buf[(t+1)&1] free

    // layer 1: DMA next step's xp tile while this step computes
    if (layer == 1 && w == 0 && (t + 1) < T_LEN) {
      tdm_load_f32_2d(lds_addr_of(&sX[(t + 1) & 1][0]),
                      xp + ((long)b0 * T_LEN + (t + 1)) * HDIM, HDIM, 16,
                      (unsigned)(T_LEN * HDIM));
    }

    // overlap: stream out previous step's h while doing this step's WMMAs
    if (t > 0) copy_h(sH, hseq, b0, t - 1, tid);

    // ---- WMMA phase: acc = h @ W_hh^T  (K=256 -> 8 chained wmma per tile)
    v8f acc0 = {0.f,0.f,0.f,0.f,0.f,0.f,0.f,0.f};
    v8f acc1 = {0.f,0.f,0.f,0.f,0.f,0.f,0.f,0.f};
    const __bf16* rowp = sH + lm * LROW;
#pragma unroll
    for (int c = 0; c < 8; ++c) {
      v16bf a = lds_loadA(rowp, c * 32 + l16 * 8);
      acc0 = wmma_bf16(a, Breg[0][c], acc0);
      acc1 = wmma_bf16(a, Breg[1][c], acc1);
    }

    // layer 1: tile t must have landed (in-order TENSORcnt; t+1 outstanding)
    if (layer == 1 && w == 0) {
      if ((t + 1) < T_LEN) __builtin_amdgcn_s_wait_tensorcnt(1);
      else                 __builtin_amdgcn_s_wait_tensorcnt(0);
    }
    __syncthreads();   // barrier B: sH reads done; xp tile visible to all

    // ---- epilogue: h_new = tanh(acc + xp); write bf16 state
    const float* xbuf = (layer == 0) ? sX[0] : sX[t & 1];
#pragma unroll
    for (int v = 0; v < 8; ++v) {
      int m = v + l16 * 8;                 // C/D layout: M = v + 8*(lane/16)
      float x0, x1;
      if (layer == 0) {
        float xin = xbuf[m * 256 + (t & 255)];
        x0 = xin * wihv[0] + biasv[0];
        x1 = xin * wihv[1] + biasv[1];
      } else {
        x0 = xbuf[m * 256 + w * 32 + lm];
        x1 = xbuf[m * 256 + w * 32 + 16 + lm];
      }
      float hn0 = tanhf(acc0[v] + x0);
      float hn1 = tanhf(acc1[v] + x1);
      sH[m * LROW + w * 32 + lm]      = (__bf16)hn0;
      sH[m * LROW + w * 32 + 16 + lm] = (__bf16)hn1;
    }
  }
  __syncthreads();
  copy_h(sH, hseq, b0, T_LEN - 1, tid);
}

// ---------------------------------------------------------------------------
// Kernel 2: xp1 = h1 @ W_ih1^T + (b_ih1+b_hh1).  (128000 x 256) x (256 x 256)
// 2000 WGs; each WG: 64 rows x 256 cols. A block (contiguous 32KB) is DMA'd
// into LDS by the TDM with hardware row-padding to the LROW pitch.
// ---------------------------------------------------------------------------
__global__ __launch_bounds__(256) void xp1_gemm_kernel(
    const __bf16* __restrict__ A,       // h1 (B*T, H) bf16
    const float*  __restrict__ Wih,     // (H,H) f32
    const float*  __restrict__ ba,
    const float*  __restrict__ bb,
    float* __restrict__ out)            // xp1 (B*T, H) f32
{
  __shared__ __bf16 sA[64 * LROW];

  const int tid  = threadIdx.x;
  const int lane = tid & 31;
  const int w    = tid >> 5;
  const int lm   = lane & 15;
  const int l16  = lane >> 4;
  const long r0  = (long)blockIdx.x * 64;

  // TDM: 64*256 bf16 contiguous -> LDS with 4-DWORD pad every 128 DWORDs
  if (w == 0) {
    tdm_load_bf16_padded(lds_addr_of(&sA[0]), A + r0 * HDIM, 64 * HDIM);
    __builtin_amdgcn_s_wait_tensorcnt(0);
  }
  __syncthreads();

  const int mt = (w & 3) * 16;          // 4 M-tiles
  const int nh = (w >> 2) * 128;        // 2 N-halves of 128
  v8f acc[8];
#pragma unroll
  for (int j = 0; j < 8; ++j) acc[j] = (v8f){0.f,0.f,0.f,0.f,0.f,0.f,0.f,0.f};

  const __bf16* rowp = sA + (mt + lm) * LROW;
#pragma unroll
  for (int c = 0; c < 8; ++c) {
    v16bf a = lds_loadA(rowp, c * 32 + l16 * 8);
#pragma unroll
    for (int j = 0; j < 8; ++j) {
      int n = nh + j * 16 + lm;
      const float4* p =
          (const float4*)(Wih + (long)n * HDIM + c * 32 + l16 * 8);
      acc[j] = wmma_bf16(a, cvt16(p), acc[j]);
    }
  }

#pragma unroll
  for (int j = 0; j < 8; ++j) {
    int n = nh + j * 16 + lm;
    float bias = ba[n] + bb[n];
#pragma unroll
    for (int v = 0; v < 8; ++v) {
      long r = r0 + mt + v + l16 * 8;
      out[r * HDIM + n] = acc[j][v] + bias;
    }
  }
}

// ---------------------------------------------------------------------------
// Kernel 4: fc1 split-K.  z_part[chunk] = feats(128 x 8000-chunk) @ fc1_w^T
// 1024 waves: wave = (chunk(32), mg(4), ng(8)); each 32Mx64N, K=8000.
// ---------------------------------------------------------------------------
__global__ __launch_bounds__(256) void fc1_kernel(
    const __bf16* __restrict__ A,       // feats (128, 256000) bf16
    const float*  __restrict__ Wf,      // (512, 256000) f32
    float* __restrict__ zp)             // (KSPLIT, 128, 512) f32 partials
{
  const int tid  = threadIdx.x;
  const int lane = tid & 31;
  const int w    = tid >> 5;
  const int lm   = lane & 15;
  const int l16  = lane >> 4;

  const int W_    = blockIdx.x * 8 + w;   // 0..1023
  const int chunk = W_ >> 5;              // 0..31
  const int rem   = W_ & 31;
  const int mt    = (rem & 3) * 32;       // 0..96
  const int nt    = (rem >> 2) * 64;      // 0..448
  const long k0   = (long)chunk * KCHUNK;

  const __bf16* pa0 = A + (long)(mt + lm)      * FC_K + k0 + l16 * 8;
  const __bf16* pa1 = A + (long)(mt + 16 + lm) * FC_K + k0 + l16 * 8;
  const float*  pb0 = Wf + (long)(nt + lm)      * FC_K + k0 + l16 * 8;
  const float*  pb1 = Wf + (long)(nt + 16 + lm) * FC_K + k0 + l16 * 8;
  const float*  pb2 = Wf + (long)(nt + 32 + lm) * FC_K + k0 + l16 * 8;
  const float*  pb3 = Wf + (long)(nt + 48 + lm) * FC_K + k0 + l16 * 8;

  v8f acc0[4], acc1[4];
#pragma unroll
  for (int j = 0; j < 4; ++j) {
    acc0[j] = (v8f){0.f,0.f,0.f,0.f,0.f,0.f,0.f,0.f};
    acc1[j] = (v8f){0.f,0.f,0.f,0.f,0.f,0.f,0.f,0.f};
  }

  for (int kc = 0; kc < KCHUNK / 32; ++kc) {
    __builtin_prefetch(pb0 + 2048, 0, 1);   // stream 512MB weight ahead
    v16bf a0 = mk16(*(const v8bf*)pa0, *(const v8bf*)(pa0 + 16));
    v16bf a1 = mk16(*(const v8bf*)pa1, *(const v8bf*)(pa1 + 16));
    v16bf b0 = cvt16((const float4*)pb0);
    v16bf b1 = cvt16((const float4*)pb1);
    v16bf b2 = cvt16((const float4*)pb2);
    v16bf b3 = cvt16((const float4*)pb3);
    acc0[0] = wmma_bf16(a0, b0, acc0[0]);  acc1[0] = wmma_bf16(a1, b0, acc1[0]);
    acc0[1] = wmma_bf16(a0, b1, acc0[1]);  acc1[1] = wmma_bf16(a1, b1, acc1[1]);
    acc0[2] = wmma_bf16(a0, b2, acc0[2]);  acc1[2] = wmma_bf16(a1, b2, acc1[2]);
    acc0[3] = wmma_bf16(a0, b3, acc0[3]);  acc1[3] = wmma_bf16(a1, b3, acc1[3]);
    pa0 += 32; pa1 += 32; pb0 += 32; pb1 += 32; pb2 += 32; pb3 += 32;
  }

  float* zb = zp + (long)chunk * (128 * 512);
#pragma unroll
  for (int j = 0; j < 4; ++j) {
    int n = nt + j * 16 + lm;
#pragma unroll
    for (int v = 0; v < 8; ++v) {
      int m0 = mt + v + l16 * 8;
      zb[(long)m0 * 512 + n]        = acc0[j][v];
      zb[(long)(m0 + 16) * 512 + n] = acc1[j][v];
    }
  }
}

// Deterministic split-K reduction: z = fc1_b + sum_chunks z_part
__global__ __launch_bounds__(256) void fc1_reduce_kernel(
    const float* __restrict__ zp, const float* __restrict__ fc1b,
    float* __restrict__ z)
{
  int idx = blockIdx.x * 256 + threadIdx.x;      // 0..65535
  float s = fc1b[idx & 511];
#pragma unroll 4
  for (int c = 0; c < KSPLIT; ++c) s += zp[(long)c * 65536 + idx];
  z[idx] = s;
}

// ---------------------------------------------------------------------------
// Kernel 6: head — LN(512) -> ReLU -> fc2(512->4) -> LN(4). One block per row.
// ---------------------------------------------------------------------------
__global__ __launch_bounds__(256) void head_kernel(
    const float* __restrict__ z,
    const float* __restrict__ ln1w, const float* __restrict__ ln1b,
    const float* __restrict__ fc2w, const float* __restrict__ fc2b,
    const float* __restrict__ ln2w, const float* __restrict__ ln2b,
    float* __restrict__ out)
{
  __shared__ float red[256];
  __shared__ float o4[4];
  const int b = blockIdx.x, tid = threadIdx.x;

  float x0 = z[b * 512 + tid];
  float x1 = z[b * 512 + 256 + tid];

  red[tid] = x0 + x1;
  __syncthreads();
  for (int s = 128; s > 0; s >>= 1) {
    if (tid < s) red[tid] += red[tid + s];
    __syncthreads();
  }
  float mu = red[0] * (1.f / 512.f);
  __syncthreads();

  float d0 = x0 - mu, d1 = x1 - mu;
  red[tid] = d0 * d0 + d1 * d1;
  __syncthreads();
  for (int s = 128; s > 0; s >>= 1) {
    if (tid < s) red[tid] += red[tid + s];
    __syncthreads();
  }
  float rs = rsqrtf(red[0] * (1.f / 512.f) + 1e-5f);
  __syncthreads();

  float y0 = fmaxf(d0 * rs * ln1w[tid] + ln1b[tid], 0.f);
  float y1 = fmaxf(d1 * rs * ln1w[tid + 256] + ln1b[tid + 256], 0.f);

  for (int j = 0; j < 4; ++j) {
    red[tid] = y0 * fc2w[j * 512 + tid] + y1 * fc2w[j * 512 + 256 + tid];
    __syncthreads();
    for (int s = 128; s > 0; s >>= 1) {
      if (tid < s) red[tid] += red[tid + s];
      __syncthreads();
    }
    if (tid == 0) o4[j] = red[0] + fc2b[j];
    __syncthreads();
  }

  if (tid == 0) {
    float m2 = (o4[0] + o4[1] + o4[2] + o4[3]) * 0.25f;
    float v2 = 0.f;
    for (int j = 0; j < 4; ++j) { float d = o4[j] - m2; v2 += d * d; }
    float r2 = rsqrtf(v2 * 0.25f + 1e-5f);
    for (int j = 0; j < 4; ++j)
      out[b * 4 + j] = (o4[j] - m2) * r2 * ln2w[j] + ln2b[j];
  }
}

// ---------------------------------------------------------------------------
// Launch
// ---------------------------------------------------------------------------
extern "C" void kernel_launch(void* const* d_in, const int* in_sizes, int n_in,
                              void* d_out, int out_size, void* d_ws, size_t ws_size,
                              hipStream_t stream) {
  const float* h0     = (const float*)d_in[0];   // (2,128,256)
  const float* input  = (const float*)d_in[1];   // (128,1000)
  const float* W_ih0  = (const float*)d_in[2];   // (256,1)
  const float* W_hh0  = (const float*)d_in[3];   // (256,256)
  const float* b_ih0  = (const float*)d_in[4];
  const float* b_hh0  = (const float*)d_in[5];
  const float* W_ih1  = (const float*)d_in[6];   // (256,256)
  const float* W_hh1  = (const float*)d_in[7];
  const float* b_ih1  = (const float*)d_in[8];
  const float* b_hh1  = (const float*)d_in[9];
  const float* fc1_w  = (const float*)d_in[10];  // (512,256000)
  const float* fc1_b  = (const float*)d_in[11];
  const float* ln1_w  = (const float*)d_in[12];
  const float* ln1_b  = (const float*)d_in[13];
  const float* fc2_w  = (const float*)d_in[14];  // (4,512)
  const float* fc2_b  = (const float*)d_in[15];
  const float* ln2_w  = (const float*)d_in[16];
  const float* ln2_b  = (const float*)d_in[17];
  float* out = (float*)d_out;                    // (128,4)

  // workspace layout
  char* ws = (char*)d_ws;
  __bf16* h1_bf  = (__bf16*)(ws);                         //  65,536,000 B
  __bf16* feats  = (__bf16*)(ws + 65536000);              //  65,536,000 B
  float*  xp1    = (float*) (ws + 131072000);             // 131,072,000 B
  float*  z_part = (float*) (ws + 262144000);             //   8,388,608 B
  float*  z      = (float*) (ws + 270532608);             //     262,144 B

  rnn_layer_kernel<<<8, 256, 0, stream>>>(
      W_hh0, h0, input, W_ih0, b_ih0, b_hh0, nullptr, h1_bf, 0);

  xp1_gemm_kernel<<<(BATCH * T_LEN) / 64, 256, 0, stream>>>(
      h1_bf, W_ih1, b_ih1, b_hh1, xp1);

  rnn_layer_kernel<<<8, 256, 0, stream>>>(
      W_hh1, h0 + BATCH * HDIM, nullptr, nullptr, nullptr, nullptr,
      xp1, feats, 1);

  fc1_kernel<<<128, 256, 0, stream>>>(feats, fc1_w, z_part);
  fc1_reduce_kernel<<<256, 256, 0, stream>>>(z_part, fc1_b, z);

  head_kernel<<<BATCH, 256, 0, stream>>>(
      z, ln1_w, ln1_b, fc2_w, fc2_b, ln2_w, ln2_b, out);
}
